// PrimitiveMixerBlock_15582141350573
// MI455X (gfx1250) — compile-verified, matrix operand order checked by reading, not run on hardware
//
#include <hip/hip_runtime.h>

typedef __bf16 bf16;
typedef __bf16 v16bf __attribute__((ext_vector_type(16)));
typedef __bf16 v8bf  __attribute__((ext_vector_type(8)));
typedef float  v8f   __attribute__((ext_vector_type(8)));

#define DMODEL  1024
#define SLEN    4096
#define BATCH   8
#define MTOT    (BATCH * SLEN)   /* 32768 rows of x */
#define NTOT    (2 * DMODEL)     /* 2048 fused output dims */
#define SCAN_BLOCKS 16

/* ---------------- workspace layout (bytes) ---------------- */
#define OFF_XBF    ((size_t)0)                               /* 32768*1024 bf16  = 64 MB  */
#define OFF_WINBF  (OFF_XBF    + (size_t)MTOT * DMODEL * 2)  /* 2048*1024 bf16   = 4 MB   */
#define OFF_WSBF   (OFF_WINBF  + (size_t)NTOT * DMODEL * 2)  /* 1024*1024 bf16   = 2 MB   */
#define OFF_G      (OFF_WSBF   + (size_t)DMODEL * DMODEL * 2)/* [S][B][D] f32    = 128 MB */
#define OFF_P      (OFF_G      + (size_t)SLEN * BATCH * DMODEL * 4)
#define OFF_SF32   (OFF_P      + (size_t)SLEN * BATCH * DMODEL * 4) /* 16*1024 f32 */
#define OFF_SBF    (OFF_SF32   + (size_t)16 * DMODEL * 4)    /* 2 slots * 16*1024 bf16 */
#define OFF_BAR    (OFF_SBF    + (size_t)2 * 16 * DMODEL * 2)
#define ZERO_BYTES ((size_t)16 * DMODEL * 4 + (size_t)2 * 16 * DMODEL * 2 + 256)

/* ---- CDNA5 async global->LDS copy (16B per lane), tracked by ASYNCcnt ---- */
__device__ __forceinline__ void async_copy16(void* lds_dst, const void* gsrc) {
  /* generic LDS pointer truncated to 32 bits == LDS byte offset (aperture rule) */
  unsigned lds = (unsigned)(size_t)lds_dst;
  asm volatile("global_load_async_to_lds_b128 %0, %1, off"
               :: "v"(lds), "v"(gsrc)
               : "memory");
}
__device__ __forceinline__ void wait_async0() {
  asm volatile("s_wait_asynccnt 0x0" ::: "memory");
}

/* ---------------- kernel 0: fp32 -> bf16 staging ---------------- */
__global__ void convert_kernel(const float* __restrict__ x,
                               const float* __restrict__ Win,
                               const float* __restrict__ Ws,
                               bf16* __restrict__ xbf,
                               bf16* __restrict__ winbf,
                               bf16* __restrict__ wsbf) {
  size_t stride = (size_t)gridDim.x * blockDim.x;
  size_t i0 = (size_t)blockIdx.x * blockDim.x + threadIdx.x;
  for (size_t i = i0; i < (size_t)MTOT * DMODEL; i += stride) xbf[i]  = (bf16)x[i];
  for (size_t i = i0; i < (size_t)NTOT * DMODEL; i += stride) winbf[i] = (bf16)Win[i];
  for (size_t i = i0; i < (size_t)DMODEL * DMODEL; i += stride) wsbf[i] = (bf16)Ws[i];
}

/* ---------------- kernel 1: z = x @ W_in^T + b_in ----------------
   gates = sigmoid(z[:, :D]) and p = z[:, D:], stored time-major [S][B][D].
   Block tile 64(M) x 256(N), K-chunks of 32, bf16 WMMA.
   8 waves in a 2x4 grid; each wave owns a 32x64 strip = 8 accumulators,
   so 8 WMMAs per 12 fragment ds_loads per K-chunk.
   Tiles staged with global_load_async_to_lds_b128 (ASYNCcnt).            */
__global__ __launch_bounds__(256)
void input_gemm_kernel(const bf16* __restrict__ xbf,
                       const bf16* __restrict__ winbf,
                       const float* __restrict__ b_in,
                       float* __restrict__ g_ws,
                       float* __restrict__ p_ws) {
  __shared__ bf16 lds_a[64 * 32];    /*  4 KB */
  __shared__ bf16 lds_b[256 * 32];   /* 16 KB */

  const int tid   = threadIdx.x;
  const int lane  = tid & 31;
  const int wave  = tid >> 5;
  const int mBase = blockIdx.y * 64;
  const int nBase = blockIdx.x * 256;

  const int wr = wave & 1;   /* M 32-half: rows wr*32 .. +32   */
  const int wc = wave >> 1;  /* N 64-chunk: cols wc*64 .. +64  */

  v8f acc[2][4];
#pragma unroll
  for (int i = 0; i < 2; ++i)
#pragma unroll
    for (int j = 0; j < 4; ++j) acc[i][j] = (v8f){};

  const int rowA = tid >> 2;         /* 0..63  */
  const int colA = (tid & 3) * 8;    /* 0,8,16,24 */
  const int lane15 = lane & 15;
  const int kh     = (lane >> 4) * 16;

  for (int kc = 0; kc < DMODEL; kc += 32) {
    /* stage A tile: 64x32 bf16, one 16B async per thread */
    async_copy16(lds_a + rowA * 32 + colA,
                 xbf + (size_t)(mBase + rowA) * DMODEL + kc + colA);
    /* stage B tile: 256x32 bf16, four 16B asyncs per thread */
#pragma unroll
    for (int i = 0; i < 4; ++i) {
      int rowB = (tid >> 2) + i * 64;
      async_copy16(lds_b + rowB * 32 + colA,
                   winbf + (size_t)(nBase + rowB) * DMODEL + kc + colA);
    }
    if (kc + 32 < DMODEL) {
      __builtin_prefetch(xbf + (size_t)(mBase + rowA) * DMODEL + kc + 32 + colA, 0, 1);
      __builtin_prefetch(winbf + (size_t)(nBase + rowA) * DMODEL + kc + 32 + colA, 0, 1);
    }
    wait_async0();
    __syncthreads();

    v16bf a0 = *(const v16bf*)(lds_a + (wr * 32 + lane15) * 32 + kh);
    v16bf a1 = *(const v16bf*)(lds_a + (wr * 32 + 16 + lane15) * 32 + kh);
#pragma unroll
    for (int sub = 0; sub < 4; ++sub) {
      v16bf b = *(const v16bf*)(lds_b + ((wc * 4 + sub) * 16 + lane15) * 32 + kh);
      acc[0][sub] = __builtin_amdgcn_wmma_f32_16x16x32_bf16(false, a0, false, b,
                                                            (short)0, acc[0][sub], false, false);
      acc[1][sub] = __builtin_amdgcn_wmma_f32_16x16x32_bf16(false, a1, false, b,
                                                            (short)0, acc[1][sub], false, false);
    }
    __syncthreads();
  }

  /* C layout: VGPR j -> M = j + 8*(lane>=16), N = lane&15 */
  const int mhalf = (lane >> 4) * 8;
#pragma unroll
  for (int mi = 0; mi < 2; ++mi) {
#pragma unroll
    for (int sub = 0; sub < 4; ++sub) {
      int col = nBase + (wc * 4 + sub) * 16 + lane15;
      float bias = b_in[col];
#pragma unroll
      for (int j = 0; j < 8; ++j) {
        int m = mBase + wr * 32 + mi * 16 + mhalf + j;
        int b = m >> 12;          /* / SLEN */
        int s = m & (SLEN - 1);
        float z = acc[mi][sub][j] + bias;
        size_t idx = ((size_t)s * BATCH + b) * DMODEL;
        if (col < DMODEL) {
          g_ws[idx + col] = 1.0f / (1.0f + __expf(-z));
        } else {
          p_ws[idx + (col - DMODEL)] = z;
        }
      }
    }
  }
}

/* ---------------- kernel 2: persistent sequential scan ----------------
   16 blocks x 128 threads (4 waves). Each wave owns 16 output dims; its
   64x1024 bf16 W_s slice lives in LDS (128 KB of 320 KB/WGP) for all 4096
   steps. State: fp32 master (exclusive per owner wave), bf16 double-
   buffered copy in global (L2-resident, 32 KB/slot) read by every wave as
   the WMMA A-matrix. Device-wide step barrier via monotonic atomic counter. */
__global__ __launch_bounds__(128)
void scan_kernel(const bf16* __restrict__ wsbf,
                 const float* __restrict__ b_s,
                 const float* __restrict__ g_ws,
                 const float* __restrict__ p_ws,
                 float* __restrict__ state_f32,
                 bf16* __restrict__ state_bf,
                 unsigned* __restrict__ barrier_cnt,
                 float* __restrict__ out) {
  extern __shared__ bf16 wlds[]; /* 64 * 1024 bf16 = 128 KB */
  const int tid     = threadIdx.x;
  const int lane    = tid & 31;
  const int wave    = tid >> 5;
  const int dimBase = blockIdx.x * 64;

  /* one-time W_s slice load into LDS via async copies */
  for (int c = tid; c < (64 * DMODEL) / 8; c += 128) {
    int e = c * 8;
    async_copy16(wlds + e, wsbf + (size_t)dimBase * DMODEL + e);
  }
  wait_async0();
  __syncthreads();

  const int kh    = (lane >> 4) * 16;
  const int lane15 = lane & 15;
  const int nloc  = wave * 16 + lane15;
  const int col   = dimBase + nloc;
  const float bias = b_s[col];
  const int mhalf = (lane >> 4) * 8;

  for (int t = 0; t < SLEN; ++t) {
    const bf16* sread  = state_bf + (size_t)(t & 1) * 16 * DMODEL;
    bf16*       swrite = state_bf + (size_t)((t + 1) & 1) * 16 * DMODEL;

    v8f acc = {};
    for (int kc = 0; kc < DMODEL; kc += 32) {
      v16bf a = *(const v16bf*)(sread + lane15 * DMODEL + kc + kh);
      v16bf b = *(const v16bf*)(wlds + nloc * DMODEL + kc + kh);
      acc = __builtin_amdgcn_wmma_f32_16x16x32_bf16(false, a, false, b,
                                                    (short)0, acc, false, false);
    }

    size_t tb = (size_t)t * BATCH * DMODEL;
#pragma unroll
    for (int j = 0; j < 8; ++j) {
      int m = mhalf + j;
      if (m < BATCH) {
        float mix  = acc[j] + bias + p_ws[tb + (size_t)m * DMODEL + col];
        float g    = g_ws[tb + (size_t)m * DMODEL + col];
        float sold = state_f32[m * DMODEL + col];
        float nxt  = g * mix + (1.0f - g) * sold;
        state_f32[m * DMODEL + col] = nxt;
        swrite[m * DMODEL + col]    = (bf16)nxt;
        out[((size_t)m * SLEN + t) * DMODEL + col] = nxt;
      }
    }

    /* device-wide step barrier (monotonic counter; reset by memset each launch) */
    __syncthreads();
    if (tid == 0) {
      __threadfence();
      atomicAdd(barrier_cnt, 1u);
      unsigned tgt = (unsigned)(t + 1) * SCAN_BLOCKS;
      while (__hip_atomic_load(barrier_cnt, __ATOMIC_ACQUIRE, __HIP_MEMORY_SCOPE_AGENT) < tgt) {
        __builtin_amdgcn_s_sleep(1);
      }
    }
    __syncthreads();
  }
}

extern "C" void kernel_launch(void* const* d_in, const int* in_sizes, int n_in,
                              void* d_out, int out_size, void* d_ws, size_t ws_size,
                              hipStream_t stream) {
  (void)in_sizes; (void)n_in; (void)out_size; (void)ws_size;
  const float* x    = (const float*)d_in[0];
  const float* W_in = (const float*)d_in[1];
  const float* b_in = (const float*)d_in[2];
  const float* W_s  = (const float*)d_in[3];
  const float* b_s  = (const float*)d_in[4];
  float* out = (float*)d_out;

  char* ws = (char*)d_ws;
  bf16*  xbf       = (bf16*)(ws + OFF_XBF);
  bf16*  winbf     = (bf16*)(ws + OFF_WINBF);
  bf16*  wsbf      = (bf16*)(ws + OFF_WSBF);
  float* g_ws      = (float*)(ws + OFF_G);
  float* p_ws      = (float*)(ws + OFF_P);
  float* state_f32 = (float*)(ws + OFF_SF32);
  bf16*  state_bf  = (bf16*)(ws + OFF_SBF);
  unsigned* bar    = (unsigned*)(ws + OFF_BAR);

  /* zero state + barrier (capture-legal) */
  hipMemsetAsync(ws + OFF_SF32, 0, ZERO_BYTES, stream);

  convert_kernel<<<2048, 256, 0, stream>>>(x, W_in, W_s, xbf, winbf, wsbf);

  input_gemm_kernel<<<dim3(NTOT / 256, MTOT / 64), 256, 0, stream>>>(
      xbf, winbf, b_in, g_ws, p_ws);

  scan_kernel<<<SCAN_BLOCKS, 128, 64 * DMODEL * sizeof(bf16), stream>>>(
      wsbf, b_s, g_ws, p_ws, state_f32, state_bf, bar, out);
}
